// SentimentGAT_34514357190729
// MI455X (gfx1250) — compile-verified, matrix operand order checked by reading, not run on hardware
//
#include <hip/hip_runtime.h>
#include <hip/hip_bf16.h>
#include <math.h>

// ---------------------------------------------------------------------------
// SentimentGAT for MI455X (gfx1250, wave32, WMMA).
//
// Dominant work: X[30000,5000] @ W1[5000,256] (76.8 GFLOP, X = 600MB fp32).
// Strategy: bf16 WMMA (v_wmma_f32_16x16x32_bf16) with on-the-fly fp32->bf16
// conversion; 128-col block tiles so X is streamed from HBM only twice;
// W tiles stay in the 192MB L2. Edge softmax uses order-preserving u32 keys
// + atomicMax, then atomicAdd scatter.
//
// v4: GEMM additionally templated on N (256 / 64 at the call sites), turning
// all B-staging and C-store strides into compile-time constants: staging is
// one base pointer per thread advanced by 32*N per k-step with per-element
// offsets folded into global_load immediates — no v_mul_u64 / s_wait_xcnt
// address recompute chains left in the 157-iteration hot loop.
// ---------------------------------------------------------------------------

typedef __bf16 bf16_t;
typedef __attribute__((ext_vector_type(16))) __bf16 v16bf;
typedef __attribute__((ext_vector_type(8)))  __bf16 v8bf;
typedef __attribute__((ext_vector_type(8)))  float  v8f;

#define NEG_SLOPE 0.2f
#define NEG_KEY   0x007FFFFFu   // order-key of -inf

// ---------------- order-preserving float <-> u32 key (for atomic max) ------
__device__ __forceinline__ unsigned fkey(float x) {
  unsigned b = __float_as_uint(x);
  return (b & 0x80000000u) ? ~b : (b | 0x80000000u);
}
__device__ __forceinline__ float funkey(unsigned u) {
  unsigned b = (u & 0x80000000u) ? (u & 0x7FFFFFFFu) : ~u;
  return __uint_as_float(b);
}

// ---------------------------------------------------------------------------
// GEMM: C[M,NC] = A[M,K] (fp32) * B[K,NC] (fp32), bf16 WMMA math, f32 accum.
// Block = 256 threads = 8 waves. Block tile: 128 rows x BN cols.
// Each wave: one 16-row stripe x (BN/16) 16x16 WMMA tiles.
// Compile-time: NC (matrix width), BN (block cols), NC % BN == 0.
// Runtime requirements: K % 4 == 0, M % 16 == 0.
// ---------------------------------------------------------------------------
#define GEMM_BM 128

template <int BN, int NC>
__global__ __launch_bounds__(256)
void gemm_bf16_wmma(const float* __restrict__ A, const float* __restrict__ B,
                    float* __restrict__ C, int M, int K)
{
  constexpr int NT  = BN / 16;              // WMMA col-tiles per wave
  constexpr int LPT = BN * 32 / 256;        // staged elements per thread
  constexpr int KST = 256 / BN;             // k-rows covered per staging pass
  __shared__ bf16_t Bs[BN][32];             // [n][k] transposed slab

  const int lane  = threadIdx.x & 31;
  const int wave  = threadIdx.x >> 5;
  const int hpart = lane >> 4;              // 0: lanes 0-15, 1: lanes 16-31
  const int l16   = lane & 15;
  const int nblk  = blockIdx.x * BN;
  const int m0    = blockIdx.y * GEMM_BM + wave * 16;
  const bool rowActive = (m0 < M);          // wave-uniform (M % 16 == 0)
  const int row   = m0 + l16;
  const float* arow = A + (long)row * K;

  // per-thread staging geometry: nn constant, kk = kbase + it*KST
  const int nn    = threadIdx.x % BN;
  const int kbase = threadIdx.x / BN;
  const float* bp = B + (long)kbase * NC + (nblk + nn);   // advance 32*NC/step
  bf16_t* lp      = &Bs[nn][kbase];                       // LDS store base

  v8f acc[NT];
#pragma unroll
  for (int j = 0; j < NT; ++j)
#pragma unroll
    for (int i = 0; i < 8; ++i) acc[j][i] = 0.0f;

  const int Kfull = K & ~31;                // full 32-wide k-steps

  // ======================= main loop: no guards anywhere ===================
  for (int k0 = 0; k0 < Kfull; k0 += 32, bp += 32 * NC) {
    // phase 1: batch-issue coalesced loads (immediate offsets off one base)
    float r[LPT];
#pragma unroll
    for (int it = 0; it < LPT; ++it)
      r[it] = bp[it * (KST * NC)];
    // phase 2: convert + transpose-scatter into LDS (immediate offsets)
#pragma unroll
    for (int it = 0; it < LPT; ++it)
      lp[it * KST] = (bf16_t)r[it];
    __syncthreads();

    if (rowActive) {
      // A fragment: 16x32 bf16 layout.
      // lanes 0-15 : K = [0..7],[16..23];  lanes 16-31: K = [8..15],[24..31]
      const int kb1 = k0 + hpart * 8;
      const float4* p1 = (const float4*)(arow + kb1);
      const float4* p2 = (const float4*)(arow + kb1 + 16);
      float4 f0 = p1[0], f1 = p1[1], f2 = p2[0], f3 = p2[1];
      v16bf afrag;
      afrag[0]  = (bf16_t)f0.x; afrag[1]  = (bf16_t)f0.y;
      afrag[2]  = (bf16_t)f0.z; afrag[3]  = (bf16_t)f0.w;
      afrag[4]  = (bf16_t)f1.x; afrag[5]  = (bf16_t)f1.y;
      afrag[6]  = (bf16_t)f1.z; afrag[7]  = (bf16_t)f1.w;
      afrag[8]  = (bf16_t)f2.x; afrag[9]  = (bf16_t)f2.y;
      afrag[10] = (bf16_t)f2.z; afrag[11] = (bf16_t)f2.w;
      afrag[12] = (bf16_t)f3.x; afrag[13] = (bf16_t)f3.y;
      afrag[14] = (bf16_t)f3.z; afrag[15] = (bf16_t)f3.w;

      if (k0 + 32 < K)                      // global_prefetch_b8 of next slab
        __builtin_prefetch(arow + (k0 + 32) + hpart * 8, 0, 3);

#pragma unroll
      for (int jt = 0; jt < NT; ++jt) {
        const bf16_t* brow = &Bs[jt * 16 + l16][0];
        v8bf blo = *(const v8bf*)(brow + hpart * 8);         // ds_load_b128
        v8bf bhi = *(const v8bf*)(brow + 16 + hpart * 8);    // ds_load_b128
        v16bf bfrag;
#pragma unroll
        for (int i = 0; i < 8; ++i) { bfrag[i] = blo[i]; bfrag[8 + i] = bhi[i]; }
        acc[jt] = __builtin_amdgcn_wmma_f32_16x16x32_bf16(
            false, afrag, false, bfrag, (short)0, acc[jt], false, false);
      }
    }
    __syncthreads();
  }

  // ======================= tail step (K % 32 != 0), runs at most once ======
  if (Kfull < K) {
    const int k0 = Kfull;
#pragma unroll
    for (int it = 0; it < LPT; ++it) {
      int kk = kbase + it * KST;            // wave-uniform guard
      float v = (k0 + kk < K) ? bp[it * (KST * NC)] : 0.0f;
      lp[it * KST] = (bf16_t)v;
    }
    __syncthreads();

    if (rowActive) {
      const int kb1 = k0 + hpart * 8;
      const int kb2 = kb1 + 16;
      v16bf afrag;
#pragma unroll
      for (int i = 0; i < 8; ++i) {
        int k1 = kb1 + i, k2 = kb2 + i;
        afrag[i]     = (bf16_t)((k1 < K) ? arow[k1] : 0.0f);
        afrag[8 + i] = (bf16_t)((k2 < K) ? arow[k2] : 0.0f);
      }
#pragma unroll
      for (int jt = 0; jt < NT; ++jt) {
        const bf16_t* brow = &Bs[jt * 16 + l16][0];
        v8bf blo = *(const v8bf*)(brow + hpart * 8);
        v8bf bhi = *(const v8bf*)(brow + 16 + hpart * 8);
        v16bf bfrag;
#pragma unroll
        for (int i = 0; i < 8; ++i) { bfrag[i] = blo[i]; bfrag[8 + i] = bhi[i]; }
        acc[jt] = __builtin_amdgcn_wmma_f32_16x16x32_bf16(
            false, afrag, false, bfrag, (short)0, acc[jt], false, false);
      }
    }
    __syncthreads();
  }

  if (rowActive) {
    // C/D layout: VGPR v -> row m0 + hpart*8 + v, col = lane&15
    float* crow = C + (long)(m0 + hpart * 8) * NC + l16 + nblk;
#pragma unroll
    for (int jt = 0; jt < NT; ++jt)
#pragma unroll
      for (int v = 0; v < 8; ++v)
        crow[(long)v * NC + jt * 16] = acc[jt][v];
  }
}

// ---------------------------------------------------------------------------
// Per-node attention logits: e_src[n,h] = <h[n,h,:], a_src[h,:]>, same for dst
// ---------------------------------------------------------------------------
__global__ void gat_logits(const float* __restrict__ h,
                           const float* __restrict__ a_src,
                           const float* __restrict__ a_dst,
                           float* __restrict__ esrc, float* __restrict__ edst,
                           int n, int H, int F)
{
  int tid = blockIdx.x * blockDim.x + threadIdx.x;
  if (tid >= n * H) return;
  int node = tid / H, hh = tid - node * H;
  const float* hp = h + (long)node * H * F + hh * F;
  const float* as = a_src + hh * F;
  const float* ad = a_dst + hh * F;
  float s1 = 0.0f, s2 = 0.0f;
  for (int f = 0; f < F; ++f) { float v = hp[f]; s1 += v * as[f]; s2 += v * ad[f]; }
  esrc[tid] = s1;
  edst[tid] = s2;
}

__global__ void fill_u32(unsigned* __restrict__ p, unsigned v, int n)
{
  int i = blockIdx.x * blockDim.x + threadIdx.x;
  if (i < n) p[i] = v;
}

__device__ __forceinline__ void edge_endpoints(const int* __restrict__ ei, int E,
                                               int idx, int& s, int& d)
{
  if (idx < E) { s = ei[idx]; d = ei[E + idx]; }
  else         { s = idx - E; d = idx - E; }        // implicit self-loops
}

__device__ __forceinline__ float edge_score(const float* esrc, const float* edst,
                                            int s, int d, int hh, int H)
{
  float v = esrc[s * H + hh] + edst[d * H + hh];
  return v > 0.0f ? v : NEG_SLOPE * v;
}

// ---- pass 1: segment max over incoming edges (u32 order-key atomicMax) ----
__global__ void edge_max(const int* __restrict__ ei, int E, int TE,
                         const float* __restrict__ esrc, const float* __restrict__ edst,
                         unsigned* __restrict__ mkey, int H)
{
  int tid = blockIdx.x * blockDim.x + threadIdx.x;
  if (tid >= TE * H) return;
  int e = tid / H, hh = tid - e * H;
  int s, d; edge_endpoints(ei, E, e, s, d);
  float v = edge_score(esrc, edst, s, d, hh, H);
  atomicMax(&mkey[d * H + hh], fkey(v));
}

// ---- pass 2: segment sum of exp(e - m) ------------------------------------
__global__ void edge_expsum(const int* __restrict__ ei, int E, int TE,
                            const float* __restrict__ esrc, const float* __restrict__ edst,
                            const unsigned* __restrict__ mkey,
                            float* __restrict__ denom, int H)
{
  int tid = blockIdx.x * blockDim.x + threadIdx.x;
  if (tid >= TE * H) return;
  int e = tid / H, hh = tid - e * H;
  int s, d; edge_endpoints(ei, E, e, s, d);
  float v = edge_score(esrc, edst, s, d, hh, H);
  float m = funkey(mkey[d * H + hh]);
  atomicAdd(&denom[d * H + hh], __expf(v - m));
}

// ---- pass 3: alpha per (edge, head) ---------------------------------------
__global__ void edge_alpha(const int* __restrict__ ei, int E, int TE,
                           const float* __restrict__ esrc, const float* __restrict__ edst,
                           const unsigned* __restrict__ mkey,
                           const float* __restrict__ denom,
                           float* __restrict__ alpha, int H)
{
  int tid = blockIdx.x * blockDim.x + threadIdx.x;
  if (tid >= TE * H) return;
  int e = tid / H, hh = tid - e * H;
  int s, d; edge_endpoints(ei, E, e, s, d);
  float v = edge_score(esrc, edst, s, d, hh, H);
  float m = funkey(mkey[d * H + hh]);
  alpha[tid] = __expf(v - m) / denom[d * H + hh];
}

// ---- pass 4: agg[dst,h,f] += alpha[e,h] * h[src,h,f] ----------------------
__global__ void edge_aggregate(const int* __restrict__ ei, int E, int TE,
                               const float* __restrict__ alpha,
                               const float* __restrict__ h,
                               float* __restrict__ agg, int H, int F)
{
  long tid = (long)blockIdx.x * blockDim.x + threadIdx.x;
  int FH = H * F;
  if (tid >= (long)TE * FH) return;
  int e  = (int)(tid / FH);
  int f  = (int)(tid - (long)e * FH);
  int hh = f / F;
  int s, d; edge_endpoints(ei, E, e, s, d);
  atomicAdd(&agg[(long)d * FH + f], alpha[(long)e * H + hh] * h[(long)s * FH + f]);
}

// ---- bias + ELU (in place) ------------------------------------------------
__global__ void bias_elu(float* __restrict__ x, const float* __restrict__ b,
                         int n, int FH)
{
  long tid = (long)blockIdx.x * blockDim.x + threadIdx.x;
  if (tid >= (long)n * FH) return;
  int f = (int)(tid % FH);
  float v = x[tid] + b[f];
  x[tid] = v > 0.0f ? v : (__expf(v) - 1.0f);
}

// ---- final classifier: out[n,c] = h[n,:] . Wc[:,c] + bc[c] ----------------
__global__ void classify(const float* __restrict__ h, const float* __restrict__ Wc,
                         const float* __restrict__ bc, float* __restrict__ out,
                         int n, int F, int C)
{
  int tid = blockIdx.x * blockDim.x + threadIdx.x;
  if (tid >= n * C) return;
  int node = tid / C, c = tid - node * C;
  const float* hp = h + (long)node * F;
  float s = bc[c];
  for (int f = 0; f < F; ++f) s += hp[f] * Wc[f * C + c];
  out[tid] = s;
}

// ---------------------------------------------------------------------------
// Host-side orchestration
// ---------------------------------------------------------------------------
static inline int cdiv_i(long a, long b) { return (int)((a + b - 1) / b); }

static void gat_attention(const float* hfeat, const float* a_s, const float* a_d,
                          const int* ei, int E, int TE, int Nn, int H, int F,
                          float* esrc, float* edst, unsigned* mkey, float* denom,
                          float* alpha, float* agg, hipStream_t stream)
{
  const int FH = H * F;
  gat_logits<<<cdiv_i((long)Nn * H, 256), 256, 0, stream>>>(hfeat, a_s, a_d,
                                                            esrc, edst, Nn, H, F);
  fill_u32<<<cdiv_i((long)Nn * H, 256), 256, 0, stream>>>(mkey, NEG_KEY, Nn * H);
  hipMemsetAsync(denom, 0, (size_t)Nn * H * sizeof(float), stream);
  edge_max<<<cdiv_i((long)TE * H, 256), 256, 0, stream>>>(ei, E, TE, esrc, edst, mkey, H);
  edge_expsum<<<cdiv_i((long)TE * H, 256), 256, 0, stream>>>(ei, E, TE, esrc, edst,
                                                             mkey, denom, H);
  edge_alpha<<<cdiv_i((long)TE * H, 256), 256, 0, stream>>>(ei, E, TE, esrc, edst,
                                                            mkey, denom, alpha, H);
  hipMemsetAsync(agg, 0, (size_t)Nn * FH * sizeof(float), stream);
  edge_aggregate<<<cdiv_i((long)TE * FH, 256), 256, 0, stream>>>(ei, E, TE, alpha,
                                                                 hfeat, agg, H, F);
}

extern "C" void kernel_launch(void* const* d_in, const int* in_sizes, int n_in,
                              void* d_out, int out_size, void* d_ws, size_t ws_size,
                              hipStream_t stream)
{
  const float* x    = (const float*)d_in[0];
  const float* W1   = (const float*)d_in[1];
  const float* a1s  = (const float*)d_in[2];
  const float* a1d  = (const float*)d_in[3];
  const float* b1   = (const float*)d_in[4];
  const float* W2   = (const float*)d_in[5];
  const float* a2s  = (const float*)d_in[6];
  const float* a2d  = (const float*)d_in[7];
  const float* b2   = (const float*)d_in[8];
  const float* W3   = (const float*)d_in[9];
  const float* a3s  = (const float*)d_in[10];
  const float* a3d  = (const float*)d_in[11];
  const float* b3   = (const float*)d_in[12];
  const float* Wc   = (const float*)d_in[13];
  const float* bc   = (const float*)d_in[14];
  const int*   ei   = (const int*)d_in[15];

  const int H = 4, F = 64, FH = H * F;              // 4 heads x 64 = 256
  const int D_IN = in_sizes[1] / FH;                // 5000
  const int Nn   = in_sizes[0] / D_IN;              // 30000
  const int E    = in_sizes[15] / 2;                // 480000
  const int TE   = E + Nn;                          // + self loops
  const int C    = in_sizes[14];                    // 3

  // workspace layout (floats); total ~ 17.9M floats (~72 MB)
  float* ws = (float*)d_ws;
  size_t off = 0;
  float*    buf0  = ws + off; off += (size_t)Nn * FH;   // layer GEMM output h
  float*    buf1  = ws + off; off += (size_t)Nn * FH;   // aggregation target
  float*    esrc  = ws + off; off += (size_t)Nn * H;
  float*    edst  = ws + off; off += (size_t)Nn * H;
  unsigned* mkey  = (unsigned*)(ws + off); off += (size_t)Nn * H;
  float*    denom = ws + off; off += (size_t)Nn * H;
  float*    alpha = ws + off; off += (size_t)TE * H;

  dim3 blk(256);

  // ---------------- layer 1: K = 5000 -> 256, 4 heads ----------------------
  {
    dim3 grid(256 / 128, cdiv_i(Nn, GEMM_BM));
    gemm_bf16_wmma<128, 256><<<grid, blk, 0, stream>>>(x, W1, buf0, Nn, D_IN);
  }
  gat_attention(buf0, a1s, a1d, ei, E, TE, Nn, H, F,
                esrc, edst, mkey, denom, alpha, buf1, stream);
  bias_elu<<<cdiv_i((long)Nn * FH, 256), blk, 0, stream>>>(buf1, b1, Nn, FH);

  // ---------------- layer 2: 256 -> 256, 4 heads ---------------------------
  {
    dim3 grid(256 / 128, cdiv_i(Nn, GEMM_BM));
    gemm_bf16_wmma<128, 256><<<grid, blk, 0, stream>>>(buf1, W2, buf0, Nn, FH);
  }
  gat_attention(buf0, a2s, a2d, ei, E, TE, Nn, H, F,
                esrc, edst, mkey, denom, alpha, buf1, stream);
  bias_elu<<<cdiv_i((long)Nn * FH, 256), blk, 0, stream>>>(buf1, b2, Nn, FH);

  // ---------------- layer 3: 256 -> 64, 1 head (mean == identity) ----------
  {
    dim3 grid(64 / 64, cdiv_i(Nn, GEMM_BM));
    gemm_bf16_wmma<64, 64><<<grid, blk, 0, stream>>>(buf1, W3, buf0, Nn, FH);
  }
  gat_attention(buf0, a3s, a3d, ei, E, TE, Nn, /*H=*/1, F,
                esrc, edst, mkey, denom, alpha, buf1, stream);
  bias_elu<<<cdiv_i((long)Nn * F, 256), blk, 0, stream>>>(buf1, b3, Nn, F);

  // ---------------- classifier ---------------------------------------------
  classify<<<cdiv_i((long)Nn * C, 256), blk, 0, stream>>>(buf1, Wc, bc,
                                                          (float*)d_out, Nn, F, C);
}
